// Cross_Attention_11269994184821
// MI455X (gfx1250) — matrix operand order repaired and checked by reading, not compiled, not run on hardware
//
#include <hip/hip_runtime.h>

typedef __bf16 bf16;
typedef __attribute__((ext_vector_type(16))) __bf16 v16bf;
typedef __attribute__((ext_vector_type(8)))  float  v8f;
typedef unsigned int u32;
typedef __attribute__((ext_vector_type(4))) unsigned int u32x4;
typedef __attribute__((ext_vector_type(8))) int i32x8;
typedef __attribute__((ext_vector_type(4))) int i32x4;

#if defined(__gfx1250__) && __has_builtin(__builtin_amdgcn_tensor_load_to_lds)
#define HAVE_TDM 1
#endif

#define WMMA_BF16(a, b, c) \
    __builtin_amdgcn_wmma_f32_16x16x32_bf16(false, (a), false, (b), (short)0, (c), false, false)

// ---------------------------------------------------------------------------
// Fragment loaders (wave32): element s of lane half h holds k = 16h + s, same order for A and B.
__device__ inline v16bf load_a_frag(const bf16* s, int ld, int lane) {
    const int m = lane & 15;
    const int kh = (lane >> 4) * 16;
    v16bf a;
#pragma unroll
    for (int i = 0; i < 8; ++i) {
        const int kk = 2 * i + kh;
        a[2 * i]     = s[m * ld + kk];
        a[2 * i + 1] = s[m * ld + kk + 1];
    }
    return a;
}

// B (32x16). Source stored rows-of-N: s[n*ld + k] (column n of B = row n of src).
__device__ inline v16bf load_b_frag_nk(const bf16* s, int ld, int lane) {
    const int n = lane & 15;
    const int koff = (lane >> 4) * 16;
    v16bf b;
#pragma unroll
    for (int i = 0; i < 16; ++i) b[i] = s[n * ld + koff + i];
    return b;
}

// B (32x16) from row-major K x N source: s[k*ld + n].
__device__ inline v16bf load_b_frag_kn(const bf16* s, int ld, int lane) {
    const int n = lane & 15;
    const int koff = (lane >> 4) * 16;
    v16bf b;
#pragma unroll
    for (int i = 0; i < 16; ++i) b[i] = s[(koff + i) * ld + n];
    return b;
}

#ifdef HAVE_TDM
// ---------------------------------------------------------------------------
// Tensor Data Mover: 2D strided tile (rows x cols bf16 elems) -> LDS, compact.
// D# per CDNA5 ISA §8: group0 {count=1, lds_addr, global_addr, type=2},
// group1 {data_size=2B, tensor_dim0=cols, tensor_dim1=rows, tile_dim0=cols,
//         tile_dim1=rows, tensor_dim0_stride=row_stride}. Higher groups zero (2D).
__device__ inline void tdm_load_tile_2d(void* lds_dst, const void* gsrc,
                                        u32 rows, u32 cols, u32 row_stride) {
    const u32 lds_addr = (u32)(uintptr_t)lds_dst;           // low 32 bits = LDS offset
    const unsigned long long ga = (unsigned long long)(uintptr_t)gsrc;
    u32x4 g0;
    g0[0] = 1u;                                             // count=1 valid descriptor
    g0[1] = lds_addr;
    g0[2] = (u32)ga;                                        // global_addr[31:0]
    g0[3] = ((u32)(ga >> 32) & 0x01FFFFFFu) | (2u << 30);   // addr[56:32] | type=2
    i32x8 g1;
    g1[0] = (int)(1u << 16);                                // data_size=1 -> 2 bytes
    g1[1] = (int)((cols & 0xFFFFu) << 16);                  // tensor_dim0[15:0]
    g1[2] = (int)((cols >> 16) | ((rows & 0xFFFFu) << 16)); // td0[31:16] | td1[15:0]
    g1[3] = (int)((rows >> 16) | (cols << 16));             // td1[31:16] | tile_dim0
    g1[4] = (int)rows;                                      // tile_dim1 ; tile_dim2=0
    g1[5] = (int)row_stride;                                // tensor_dim0_stride[31:0]
    g1[6] = 0;
    g1[7] = 0;
    const i32x4 z4 = {0, 0, 0, 0};
    const i32x8 z8 = {0, 0, 0, 0, 0, 0, 0, 0};
    __builtin_amdgcn_tensor_load_to_lds(g0, g1, z4, z4, z8, 0);
}
#endif

// ---------------------------------------------------------------------------
// Generic Out[M,N] = In[M,K] @ W[N,K]^T  (bf16 WMMA, fp32 accumulate)
template <bool IN_BF16, bool FINAL>
__global__ __launch_bounds__(128, 1) void gemm_bt(
        const void* __restrict__ in_, const float* __restrict__ w,
        const float* __restrict__ bias, void* __restrict__ out_,
        int M, int N, int K) {
    __shared__ alignas(16) bf16 Xs[64 * 32];
    __shared__ alignas(16) bf16 Ws[64 * 32];

    const int t = threadIdx.x;
    const int lane = t & 31;
    const int wv = t >> 5;
    const int n0 = blockIdx.x * 64;
    const int m0 = blockIdx.y * 64;

    v8f acc[4] = {v8f{}, v8f{}, v8f{}, v8f{}};

    const int srow = t >> 1;
    const int scb = (t & 1) * 16;

    for (int k0 = 0; k0 < K; k0 += 32) {
        if (IN_BF16) {
            const bf16* in = (const bf16*)in_;
            const u32* g = (const u32*)(in + (size_t)(m0 + srow) * K + k0 + scb);
            u32* l = (u32*)(Xs + srow * 32 + scb);
#pragma unroll
            for (int i = 0; i < 8; ++i) l[i] = g[i];
        } else {
            const float* in = (const float*)in_;
            const float* g = in + (size_t)(m0 + srow) * K + k0 + scb;
#pragma unroll
            for (int i = 0; i < 16; ++i) Xs[srow * 32 + scb + i] = (bf16)g[i];
        }
        {
            const float* g = w + (size_t)(n0 + srow) * K + k0 + scb;
#pragma unroll
            for (int i = 0; i < 16; ++i) Ws[srow * 32 + scb + i] = (bf16)g[i];
        }
        __syncthreads();

        const v16bf a = load_a_frag(Xs + (16 * wv) * 32, 32, lane);
#pragma unroll
        for (int nt = 0; nt < 4; ++nt) {
            const v16bf b = load_b_frag_nk(Ws + (16 * nt) * 32, 32, lane);
            acc[nt] = WMMA_BF16(a, b, acc[nt]);
        }
        __syncthreads();
    }

    const int ri = 8 * (lane >> 4);
    const int cl = lane & 15;
#pragma unroll
    for (int nt = 0; nt < 4; ++nt) {
#pragma unroll
        for (int i = 0; i < 8; ++i) {
            const int r = m0 + 16 * wv + ri + i;
            const int c = n0 + 16 * nt + cl;
            const float v = acc[nt][i];
            if (FINAL) {
                ((float*)out_)[(size_t)r * N + c] = v + bias[c];
            } else {
                ((bf16*)out_)[(size_t)r * N + c] = (bf16)v;
            }
        }
    }
}

// ---------------------------------------------------------------------------
#define AB 8
#define AH 8
#define AN 1024
#define AC 768
#define AHD 96

__global__ __launch_bounds__(128, 1) void flash_attn(
        const bf16* __restrict__ Q, const bf16* __restrict__ Kg,
        const bf16* __restrict__ Vg, const float* __restrict__ rp,
        bf16* __restrict__ Aout) {
    __shared__ alignas(16) bf16 Ks[2][32 * AHD];   // double-buffered via TDM
    __shared__ alignas(16) bf16 Vs[2][32 * AHD];
    __shared__ alignas(16) bf16 Ps[4][16 * 32];

    const int t = threadIdx.x;
    const int lane = t & 31;
    const int wv = t >> 5;

    const int mchunk = blockIdx.x & 15;
    const int bh = blockIdx.x >> 4;
    const int h = bh & 7;
    const int b = bh >> 3;
    const int m0 = mchunk * 64;
    const int bN = b * AN;
    const int hHD = h * AHD;
    const float scale = 0.1020620726159658f;  // 96^-0.5

    const bf16* qrow = Q + (size_t)(bN + m0 + 16 * wv) * AC + hHD;
    v16bf qa[3];
#pragma unroll
    for (int s = 0; s < 3; ++s) qa[s] = load_a_frag(qrow + 32 * s, AC, lane);

    v8f o[6] = {v8f{}, v8f{}, v8f{}, v8f{}, v8f{}, v8f{}};
    float m_run[8], l_run[8];
#pragma unroll
    for (int i = 0; i < 8; ++i) { m_run[i] = -1e30f; l_run[i] = 0.0f; }

    const int ri = 8 * (lane >> 4);
    const int cl = lane & 15;
#ifndef HAVE_TDM
    const int crow = t >> 2;
    const int cseg = (t & 3) * 12;
#endif

#ifdef HAVE_TDM
    // prologue: start streaming chunk 0 (TENSORcnt tracks completion)
    if (wv == 0) {
        tdm_load_tile_2d(&Ks[0][0], Kg + (size_t)bN * AC + hHD, 32, AHD, AC);
        tdm_load_tile_2d(&Vs[0][0], Vg + (size_t)bN * AC + hHD, 32, AHD, AC);
    }
#endif

    for (int kv0 = 0; kv0 < AN; kv0 += 32) {
        const int buf = (kv0 >> 5) & 1;
#ifdef HAVE_TDM
        // issue next chunk into the other buffer, then wait for current chunk:
        // TDM ops of one wave retire in order, so TENSORcnt<=2 means the two
        // loads of chunk kv0 are complete while chunk kv0+32 streams behind.
        if (wv == 0) {
            if (kv0 + 32 < AN) {
                const size_t nbase = (size_t)(bN + kv0 + 32) * AC + hHD;
                tdm_load_tile_2d(&Ks[buf ^ 1][0], Kg + nbase, 32, AHD, AC);
                tdm_load_tile_2d(&Vs[buf ^ 1][0], Vg + nbase, 32, AHD, AC);
                __builtin_amdgcn_s_wait_tensorcnt(2);
            } else {
                __builtin_amdgcn_s_wait_tensorcnt(0);
            }
        }
#else
        {
            const u32* gk = (const u32*)(Kg + (size_t)(bN + kv0 + crow) * AC + hHD) + cseg;
            const u32* gv = (const u32*)(Vg + (size_t)(bN + kv0 + crow) * AC + hHD) + cseg;
            u32* lk = (u32*)&Ks[buf][0] + crow * 48 + cseg;
            u32* lv = (u32*)&Vs[buf][0] + crow * 48 + cseg;
#pragma unroll
            for (int i = 0; i < 12; ++i) lk[i] = gk[i];
#pragma unroll
            for (int i = 0; i < 12; ++i) lv[i] = gv[i];
        }
#endif
        __syncthreads();

        const bf16* ks = &Ks[buf][0];
        const bf16* vs = &Vs[buf][0];

        // ---- S = Q K^T : two 16x16 C tiles covering 32 kv columns ----
        v8f s0 = {}, s1 = {};
#pragma unroll
        for (int s = 0; s < 3; ++s) {
            const v16bf bk0 = load_b_frag_nk(ks + 32 * s, AHD, lane);
            const v16bf bk1 = load_b_frag_nk(ks + 16 * AHD + 32 * s, AHD, lane);
            s0 = WMMA_BF16(qa[s], bk0, s0);
            s1 = WMMA_BF16(qa[s], bk1, s1);
        }

        // ---- scale + relative-position bias ----
        const float* rpb = rp + ((size_t)h * AN + (m0 + 16 * wv)) * AN + kv0;
        if (kv0 + 32 < AN) __builtin_prefetch(rpb + 32, 0, 0);
#pragma unroll
        for (int i = 0; i < 8; ++i) {
            s0[i] = s0[i] * scale + rpb[(size_t)(ri + i) * AN + cl];
            s1[i] = s1[i] * scale + rpb[(size_t)(ri + i) * AN + 16 + cl];
        }

        // ---- online softmax ----
        float corr[8];
#pragma unroll
        for (int i = 0; i < 8; ++i) {
            float mx = fmaxf(s0[i], s1[i]);
#pragma unroll
            for (int off = 8; off; off >>= 1) mx = fmaxf(mx, __shfl_xor(mx, off, 32));
            const float mnew = fmaxf(m_run[i], mx);
            corr[i] = __expf(m_run[i] - mnew);
            s0[i] = __expf(s0[i] - mnew);
            s1[i] = __expf(s1[i] - mnew);
            float rs = s0[i] + s1[i];
#pragma unroll
            for (int off = 8; off; off >>= 1) rs += __shfl_xor(rs, off, 32);
            l_run[i] = l_run[i] * corr[i] + rs;
            m_run[i] = mnew;
        }
#pragma unroll
        for (int nt = 0; nt < 6; ++nt)
#pragma unroll
            for (int i = 0; i < 8; ++i) o[nt][i] *= corr[i];

        // ---- re-layout P: C-fragment -> LDS -> A-fragment ----
        bf16* ps = &Ps[wv][0];
#pragma unroll
        for (int i = 0; i < 8; ++i) {
            ps[(ri + i) * 32 + cl]      = (bf16)s0[i];
            ps[(ri + i) * 32 + 16 + cl] = (bf16)s1[i];
        }
        const v16bf pa = load_a_frag(ps, 32, lane);

        // ---- O += P @ V ----
#pragma unroll
        for (int nt = 0; nt < 6; ++nt) {
            const v16bf bv = load_b_frag_kn(vs + 16 * nt, AHD, lane);
            o[nt] = WMMA_BF16(pa, bv, o[nt]);
        }
        __syncthreads();
    }

    float inv[8];
#pragma unroll
    for (int i = 0; i < 8; ++i) inv[i] = 1.0f / l_run[i];
#pragma unroll
    for (int nt = 0; nt < 6; ++nt)
#pragma unroll
        for (int i = 0; i < 8; ++i) {
            const int r = bN + m0 + 16 * wv + ri + i;
            Aout[(size_t)r * AC + hHD + 16 * nt + cl] = (bf16)(o[nt][i] * inv[i]);
        }
}

// ---------------------------------------------------------------------------
extern "C" void kernel_launch(void* const* d_in, const int* in_sizes, int n_in,
                              void* d_out, int out_size, void* d_ws, size_t ws_size,
                              hipStream_t stream) {
    const float* x  = (const float*)d_in[0];
    const float* y  = (const float*)d_in[1];
    const float* rp = (const float*)d_in[2];
    const float* Wq = (const float*)d_in[5];
    const float* Wk = (const float*)d_in[6];
    const float* Wv = (const float*)d_in[7];
    const float* Wp = (const float*)d_in[8];
    const float* bp = (const float*)d_in[9];
    float* out = (float*)d_out;

    const int M = AB * AN;
    const int C = AC;
    const size_t bufElems = (size_t)M * C;
    char* ws = (char*)d_ws;
    bf16* Qw = (bf16*)(ws);
    bf16* Kw = (bf16*)(ws + 2 * bufElems);
    bf16* Vw = (bf16*)(ws + 4 * bufElems);
    bf16* Aw = (bf16*)(ws + 6 * bufElems);

    const dim3 blk(128);
    const dim3 ggrid(C / 64, M / 64);

    gemm_bt<false, false><<<ggrid, blk, 0, stream>>>(x, Wq, nullptr, Qw, M, C, C);
    gemm_bt<false, false><<<ggrid, blk, 0, stream>>>(y, Wk, nullptr, Kw, M, C, C);
    gemm_bt<false, false><<<ggrid, blk, 0, stream>>>(y, Wv, nullptr, Vw, M, C, C);

    flash_attn<<<dim3(AB * AH * (AN / 64)), blk, 0, stream>>>(Qw, Kw, Vw, rp, Aw);

    gemm_bt<true, true><<<ggrid, blk, 0, stream>>>(Aw, Wp, bp, out, M, C, C);
}
